// DeltaNet_28097676051326
// MI455X (gfx1250) — matrix operand order repaired and checked by reading, not compiled, not run on hardware
//
#include <hip/hip_runtime.h>
#include <hip/hip_bf16.h>

// ---------------------------------------------------------------------------
// DeltaNet forward for MI455X (gfx1250, wave32, WMMA + async-to-LDS).
//   [K0] fp32 -> bf16 casts (x, Wq, Wk, Wv, Wb, Wo)
//   [K1] WMMA bf16 GEMMs: q/k/v = x@W^T (f32 out), beta = sigmoid(x@Wb^T)
//   [K2] depthwise causal conv + SiLU (+ L2 norm & q-scale) -> [b,h,l,64]
//   [K3] delta-rule scan: S in VGPRs, k/q/v/beta double-buffered in LDS via
//        GLOBAL_LOAD_ASYNC_TO_LDS_B128 (ASYNCcnt), fused RMSNorm -> bf16
//   [K4] WMMA bf16 GEMM: out = o_norm @ Wo^T -> d_out ; S -> d_out tail
// ---------------------------------------------------------------------------

#define HIDDEN 1024
#define NHEADS 16
#define HEAD_DIM 64
#define BATCH 4
#define SEQ 1024
#define TOKENS (BATCH * SEQ)   // 4096

typedef __attribute__((ext_vector_type(16))) __bf16 v16bf;
typedef __attribute__((ext_vector_type(8)))  float  v8f;

#if __has_builtin(__builtin_amdgcn_global_load_async_to_lds_b128) && \
    __has_builtin(__builtin_amdgcn_global_load_async_to_lds_b32) &&  \
    __has_builtin(__builtin_amdgcn_s_wait_asynccnt)
#define USE_ASYNC_LDS 1
typedef __attribute__((ext_vector_type(4))) int vi4;
typedef __attribute__((address_space(1))) vi4 as1_vi4;   // global int4
typedef __attribute__((address_space(3))) vi4 as3_vi4;   // LDS int4
typedef __attribute__((address_space(1))) int as1_i32;   // global int
typedef __attribute__((address_space(3))) int as3_i32;   // LDS int
#else
#define USE_ASYNC_LDS 0
#endif

__device__ __forceinline__ __bf16 f2bf(float f) {
  unsigned u = __float_as_uint(f);
  unsigned r = u + 0x7FFFu + ((u >> 16) & 1u);   // round-to-nearest-even
  unsigned short s = (unsigned short)(r >> 16);
  __bf16 out;
  __builtin_memcpy(&out, &s, 2);
  return out;
}

// ---------------------------------------------------------------- K0: casts
__global__ void cast_f32_bf16(const float* __restrict__ in,
                              __bf16* __restrict__ out, int n) {
  int i = blockIdx.x * blockDim.x + threadIdx.x;
  if (i < n) out[i] = f2bf(in[i]);
}

// ------------------------------------------------------- K1/K4: WMMA GEMM
// C[M,N] = A[M,K] @ W[N,K]^T  (both K-contiguous -> clean fragment loads)
// Wave tile (MT*16)x(NT*16); block = BWM x BWN waves; K compile-time.
template <int MT, int NT, int BWM, int BWN, int K, bool SIGMOID>
__global__ void gemm_bf16_wmma(const __bf16* __restrict__ A,
                               const __bf16* __restrict__ W,
                               float* __restrict__ C, int N) {
  const int lane = threadIdx.x & 31;
  const int wave = threadIdx.x >> 5;
  const int wm = wave / BWN, wn = wave % BWN;
  const int tileM0 = blockIdx.y * (BWM * MT * 16) + wm * (MT * 16);
  const int tileN0 = blockIdx.x * (BWN * NT * 16) + wn * (NT * 16);
  const int half = lane >> 4;    // 16-lane half
  const int idx  = lane & 15;    // row/col within tile

  v8f acc[MT][NT];
  #pragma unroll
  for (int i = 0; i < MT; i++)
    #pragma unroll
    for (int j = 0; j < NT; j++) acc[i][j] = (v8f)(0.0f);

  const __bf16* pa[MT];
  const __bf16* pb[NT];
  #pragma unroll
  for (int i = 0; i < MT; i++)
    pa[i] = A + (size_t)(tileM0 + i * 16 + idx) * K + half * 8;
  #pragma unroll
  for (int j = 0; j < NT; j++)
    pb[j] = W + (size_t)(tileN0 + j * 16 + idx) * K + half * 8;

  #pragma unroll 2
  for (int kb = 0; kb < K; kb += 32) {
    v16bf afrag[MT], bfrag[NT];
    // A frag (16x32 bf16): lane(half,idx): elems0..7 = K[kb+half*8 ..),
    // elems8..15 = K[kb+16+half*8 ..)
    #pragma unroll
    for (int i = 0; i < MT; i++) {
      const __bf16* p = pa[i] + kb;
      *((uint4*)&afrag[i])     = *(const uint4*)p;
      *((uint4*)&afrag[i] + 1) = *(const uint4*)(p + 16);
    }
    // B frag (32x16 bf16): same K order as A: elems0..7 = K[kb+half*8 ..), elems8..15 = K[kb+16+half*8 ..)
    #pragma unroll
    for (int j = 0; j < NT; j++) {
      const __bf16* p = pb[j] + kb;
      *((uint4*)&bfrag[j])     = *(const uint4*)p;
      *((uint4*)&bfrag[j] + 1) = *(const uint4*)(p + 16);
    }
    #pragma unroll
    for (int i = 0; i < MT; i++)
      #pragma unroll
      for (int j = 0; j < NT; j++)
        acc[i][j] = __builtin_amdgcn_wmma_f32_16x16x32_bf16(
            false, afrag[i], false, bfrag[j], (short)0, acc[i][j],
            false, false);
  }

  // C/D layout: VGPR r, lanes0-15 -> M=r, lanes16-31 -> M=8+r; N=idx.
  #pragma unroll
  for (int i = 0; i < MT; i++)
    #pragma unroll
    for (int j = 0; j < NT; j++)
      #pragma unroll
      for (int r = 0; r < 8; r++) {
        const int m = tileM0 + i * 16 + half * 8 + r;
        const int n = tileN0 + j * 16 + idx;
        float v = acc[i][j][r];
        if (SIGMOID) v = 1.0f / (1.0f + __expf(-v));
        C[(size_t)m * N + n] = v;
      }
}

// ------------------------------------- K2: causal conv + SiLU (+ L2 norm)
// mode 0: q (l2norm + *dk^-0.5), 1: k (l2norm), 2: v (plain)
__global__ void conv_silu_norm(const float* __restrict__ X,
                               const float* __restrict__ Wc,
                               float* __restrict__ OH, int mode) {
  const int gwave = (blockIdx.x * blockDim.x + threadIdx.x) >> 5;
  const int lane = threadIdx.x & 31;
  const int h = gwave & (NHEADS - 1);
  const int bl = gwave >> 4;          // b*SEQ + l
  const int l = bl & (SEQ - 1);
  const int b = bl >> 10;

  float y[2];
  #pragma unroll
  for (int t = 0; t < 2; t++) {
    const int c = h * HEAD_DIM + lane + t * 32;
    const float* w = Wc + c * 4;
    float acc = 0.0f;
    #pragma unroll
    for (int kk = 0; kk < 4; kk++) {
      const int ls = l - 3 + kk;
      if (ls >= 0) acc += X[((size_t)b * SEQ + ls) * HIDDEN + c] * w[kk];
    }
    y[t] = acc * (1.0f / (1.0f + __expf(-acc)));   // SiLU
  }
  if (mode < 2) {
    float ss = y[0] * y[0] + y[1] * y[1];
    #pragma unroll
    for (int off = 16; off > 0; off >>= 1) ss += __shfl_xor(ss, off, 32);
    float inv = 1.0f / fmaxf(sqrtf(ss), 1e-6f);
    if (mode == 0) inv *= 0.125f;                  // dk^-0.5
    y[0] *= inv;
    y[1] *= inv;
  }
  const size_t base = (((size_t)b * NHEADS + h) * SEQ + l) * HEAD_DIM;
  OH[base + lane] = y[0];
  OH[base + lane + 32] = y[1];
}

// ----------------------------------------------- K3: delta-rule scan + RMS
// One block per (b,h); 64 threads; thread m owns S[:,m] in 64 VGPRs.
// k/q/v/beta for step l+1 are prefetched into the other LDS buffer with
// GLOBAL_LOAD_ASYNC_TO_LDS_B128 while step l computes.
__global__ void delta_scan(const float* __restrict__ QH,
                           const float* __restrict__ KH,
                           const float* __restrict__ VH,
                           const float* __restrict__ BETA, // [b*l,16]
                           const float* __restrict__ ONW,  // [64]
                           __bf16* __restrict__ OBF,       // [b,l,1024] bf16
                           float* __restrict__ SOUT) {     // [b,h,64,64]
  const int bh = blockIdx.x;            // 0..63
  const int b = bh >> 4, h = bh & 15;
  const int m = threadIdx.x;            // 0..63 : dv column

  float S[HEAD_DIM];
  #pragma unroll
  for (int d = 0; d < HEAD_DIM; d++) S[d] = 0.0f;

  __shared__ float buf[2][196];         // {k[64], q[64], v[64], beta}
  __shared__ float red[2];

  const size_t base = (size_t)bh * SEQ * HEAD_DIM;
  const float onw_m = ONW[m];

#if USE_ASYNC_LDS
  const int sec = m >> 4;               // 0:k 1:q 2:v 3:(beta lane)
  const int col = (m & 15) * 4;         // 4 floats / 16B per lane
  auto issue = [&](int l, int slot) {
    if (sec < 3) {
      const float* gp;
      if (sec == 0)      gp = KH + base + (size_t)l * HEAD_DIM + col;
      else if (sec == 1) gp = QH + base + (size_t)l * HEAD_DIM + col;
      else               gp = VH + base + (size_t)l * HEAD_DIM + col;
      __builtin_amdgcn_global_load_async_to_lds_b128(
          (as1_vi4*)gp, (as3_vi4*)&buf[slot][sec * 64 + col], 0, 0);
    } else if (m == 48) {
      const float* gp = BETA + ((size_t)b * SEQ + l) * NHEADS + h;
      __builtin_amdgcn_global_load_async_to_lds_b32(
          (as1_i32*)gp, (as3_i32*)&buf[slot][192], 0, 0);
    }
  };
  issue(0, 0);
#endif

  for (int l = 0; l < SEQ; l++) {
    const int cur = l & 1;
#if USE_ASYNC_LDS
    __builtin_amdgcn_s_wait_asynccnt(0);  // buf[cur] landed (my pieces)
    __syncthreads();                      // all waves' pieces visible;
                                          // old readers of buf[!cur] done
    issue(l + 1, cur ^ 1);                // prefetch next step (overread at
                                          // l==1023 stays inside workspace)
    const float* kb = &buf[cur][0];
    const float* qb = &buf[cur][HEAD_DIM];
    const float vt = buf[cur][2 * HEAD_DIM + m];
    const float bt = buf[cur][192];
#else
    buf[cur][m]            = KH[base + (size_t)l * HEAD_DIM + m];
    buf[cur][HEAD_DIM + m] = QH[base + (size_t)l * HEAD_DIM + m];
    const float vt = VH[base + (size_t)l * HEAD_DIM + m];
    const float bt = BETA[((size_t)b * SEQ + l) * NHEADS + h];
    __syncthreads();
    const float* kb = &buf[cur][0];
    const float* qb = &buf[cur][HEAD_DIM];
#endif

    float acc = 0.0f;
    #pragma unroll
    for (int d = 0; d < HEAD_DIM; d++) acc += kb[d] * S[d];
    const float vnew = (vt - acc) * bt;

    float o = 0.0f;
    #pragma unroll
    for (int d = 0; d < HEAD_DIM; d++) {
      S[d] += kb[d] * vnew;
      o += qb[d] * S[d];
    }

    // RMSNorm across 64 columns: shfl within wave + LDS cross-wave
    float ss = o * o;
    #pragma unroll
    for (int off = 16; off > 0; off >>= 1) ss += __shfl_xor(ss, off, 32);
    if ((m & 31) == 0) red[m >> 5] = ss;
    __syncthreads();
    ss = red[0] + red[1];
    const float on = o * rsqrtf(ss * (1.0f / HEAD_DIM) + 1e-6f) * onw_m;

    OBF[((size_t)b * SEQ + l) * HIDDEN + h * HEAD_DIM + m] = f2bf(on);
  }

  #pragma unroll
  for (int d = 0; d < HEAD_DIM; d++)
    SOUT[((size_t)bh * HEAD_DIM + d) * HEAD_DIM + m] = S[d];
}

// ---------------------------------------------------------------------------
extern "C" void kernel_launch(void* const* d_in, const int* in_sizes, int n_in,
                              void* d_out, int out_size, void* d_ws,
                              size_t ws_size, hipStream_t stream) {
  const float* x   = (const float*)d_in[0];
  const float* Wq  = (const float*)d_in[1];
  const float* Wk  = (const float*)d_in[2];
  const float* Wv  = (const float*)d_in[3];
  const float* Wb  = (const float*)d_in[4];
  const float* Wo  = (const float*)d_in[5];
  const float* cq  = (const float*)d_in[6];
  const float* ck  = (const float*)d_in[7];
  const float* cv  = (const float*)d_in[8];
  const float* onw = (const float*)d_in[9];
  float* out  = (float*)d_out;                       // [4,1024,1024]
  float* Sout = out + (size_t)TOKENS * HIDDEN;       // [4,16,64,64]

  char* ws = (char*)d_ws;
  size_t off = 0;
  auto alloc = [&](size_t bytes) -> void* {
    void* p = ws + off;
    off += (bytes + 255) & ~(size_t)255;
    return p;
  };
  __bf16* xb  = (__bf16*)alloc((size_t)TOKENS * HIDDEN * 2);
  __bf16* Wqb = (__bf16*)alloc((size_t)HIDDEN * HIDDEN * 2);
  __bf16* Wkb = (__bf16*)alloc((size_t)HIDDEN * HIDDEN * 2);
  __bf16* Wvb = (__bf16*)alloc((size_t)HIDDEN * HIDDEN * 2);
  __bf16* Wbb = (__bf16*)alloc((size_t)NHEADS * HIDDEN * 2);
  __bf16* Wob = (__bf16*)alloc((size_t)HIDDEN * HIDDEN * 2);
  float* qp    = (float*)alloc((size_t)TOKENS * HIDDEN * 4);
  float* kp    = (float*)alloc((size_t)TOKENS * HIDDEN * 4);
  float* vp    = (float*)alloc((size_t)TOKENS * HIDDEN * 4);
  float* betab = (float*)alloc((size_t)TOKENS * NHEADS * 4);
  float* qh    = (float*)alloc((size_t)TOKENS * HIDDEN * 4);
  float* kh    = (float*)alloc((size_t)TOKENS * HIDDEN * 4);
  float* vh    = (float*)alloc((size_t)TOKENS * HIDDEN * 4);
  __bf16* obf = (__bf16*)alloc((size_t)TOKENS * HIDDEN * 2);
  (void)ws_size;

  // ---- K0: casts ----
  {
    int n = TOKENS * HIDDEN;
    cast_f32_bf16<<<(n + 255) / 256, 256, 0, stream>>>(x, xb, n);
    n = HIDDEN * HIDDEN;
    cast_f32_bf16<<<(n + 255) / 256, 256, 0, stream>>>(Wq, Wqb, n);
    cast_f32_bf16<<<(n + 255) / 256, 256, 0, stream>>>(Wk, Wkb, n);
    cast_f32_bf16<<<(n + 255) / 256, 256, 0, stream>>>(Wv, Wvb, n);
    cast_f32_bf16<<<(n + 255) / 256, 256, 0, stream>>>(Wo, Wob, n);
    n = NHEADS * HIDDEN;
    cast_f32_bf16<<<(n + 255) / 256, 256, 0, stream>>>(Wb, Wbb, n);
  }

  // ---- K1: projections (bf16 WMMA, f32 accumulate) ----
  {
    dim3 grid(HIDDEN / 128, TOKENS / 64);         // 8 x 64 blocks
    dim3 blk(128);                                // 4 waves: 2x2, tile 64x128
    gemm_bf16_wmma<2, 4, 2, 2, HIDDEN, false><<<grid, blk, 0, stream>>>(
        xb, Wqb, qp, HIDDEN);
    gemm_bf16_wmma<2, 4, 2, 2, HIDDEN, false><<<grid, blk, 0, stream>>>(
        xb, Wkb, kp, HIDDEN);
    gemm_bf16_wmma<2, 4, 2, 2, HIDDEN, false><<<grid, blk, 0, stream>>>(
        xb, Wvb, vp, HIDDEN);
    dim3 gridb(1, TOKENS / 16);                   // beta: N=16
    gemm_bf16_wmma<1, 1, 1, 1, HIDDEN, true><<<gridb, dim3(32), 0, stream>>>(
        xb, Wbb, betab, NHEADS);
  }

  // ---- K2: conv + SiLU (+ L2 norm) ----
  {
    const int blocks = (TOKENS * NHEADS) / 8;     // one wave32 per (b,l,h)
    conv_silu_norm<<<blocks, 256, 0, stream>>>(qp, cq, qh, 0);
    conv_silu_norm<<<blocks, 256, 0, stream>>>(kp, ck, kh, 1);
    conv_silu_norm<<<blocks, 256, 0, stream>>>(vp, cv, vh, 2);
  }

  // ---- K3: delta-rule scan + RMSNorm ----
  delta_scan<<<BATCH * NHEADS, HEAD_DIM, 0, stream>>>(qh, kh, vh, betab, onw,
                                                      obf, Sout);

  // ---- K4: output projection ----
  {
    dim3 grid(HIDDEN / 128, TOKENS / 64);
    gemm_bf16_wmma<2, 4, 2, 2, HIDDEN, false><<<grid, dim3(128), 0, stream>>>(
        obf, Wob, out, HIDDEN);
  }
}
